// NonLocalNet_2353642078228
// MI455X (gfx1250) — matrix operand order repaired and checked by reading, not compiled, not run on hardware
//
#include <hip/hip_runtime.h>

typedef __bf16 bf16;
typedef __attribute__((ext_vector_type(16))) __bf16 v16bf;
typedef __attribute__((ext_vector_type(8)))  __bf16 v8bf;
typedef __attribute__((ext_vector_type(8)))  float  v8f;

#define WPB 4
#define EPSV 1e-5f

// ---- WMMA fragment helpers (gfx1250 wave32, 16x16x32 bf16) ----
// A (16x32, M x K), row-major memory; lane: m=l&15, grp=l>>4 holds
// K = grp*8+{0..7} (elems 0..7) and 16+grp*8+{0..7} (elems 8..15).
__device__ __forceinline__ v16bf load_fragA(const bf16* row, int grp) {
  const v8bf lo = *(const v8bf*)(row + grp * 8);
  const v8bf hi = *(const v8bf*)(row + 16 + grp * 8);
  v16bf r;
#pragma unroll
  for (int i = 0; i < 8; ++i) { r[i] = lo[i]; r[8 + i] = hi[i]; }
  return r;
}
// B (32x16, K x N), stored [n][k] row-major; lane: n=l&15, grp=l>>4 holds
// K = grp*8+{0..7} (elems 0..7) and 16+grp*8+{0..7} (elems 8..15).
__device__ __forceinline__ v16bf load_fragB(const bf16* row, int grp) {
  const v8bf lo = *(const v8bf*)(row + grp * 8);
  const v8bf hi = *(const v8bf*)(row + 16 + grp * 8);
  v16bf r;
#pragma unroll
  for (int i = 0; i < 8; ++i) { r[i] = lo[i]; r[8 + i] = hi[i]; }
  return r;
}
__device__ __forceinline__ v8f wmma_bf16(v16bf a, v16bf b, v8f c) {
  return __builtin_amdgcn_wmma_f32_16x16x32_bf16(false, a, false, b, (short)0, c,
                                                 false, false);
}

// ---- 1x1 conv (64 -> 32), writes bf16 [b][p][32] ----
__global__ void k_conv1x1(const float* __restrict__ f1, const float* __restrict__ f2,
                          const float* __restrict__ w, const float* __restrict__ bias,
                          bf16* __restrict__ out, int N, int total) {
  __shared__ float ws_[32 * 64];
  __shared__ float bs_[32];
  for (int i = threadIdx.x; i < 32 * 64; i += blockDim.x) ws_[i] = w[i];
  if (threadIdx.x < 32) bs_[threadIdx.x] = bias[threadIdx.x];
  __syncthreads();
  int idx = blockIdx.x * blockDim.x + threadIdx.x;
  if (idx >= total) return;
  int b = idx / N, p = idx - b * N;
  float xv[64];
#pragma unroll
  for (int c = 0; c < 32; ++c) xv[c] = f1[((size_t)(b * 32 + c)) * N + p];
#pragma unroll
  for (int c = 0; c < 32; ++c) xv[32 + c] = f2[((size_t)(b * 32 + c)) * N + p];
  bf16* o = out + (size_t)idx * 32;
  for (int oc = 0; oc < 32; ++oc) {
    float a = bs_[oc];
    const float* wr = &ws_[oc * 64];
#pragma unroll
    for (int c = 0; c < 64; ++c) a = fmaf(wr[c], xv[c], a);
    o[oc] = (bf16)a;
  }
}

// ---- 1x1 conv + s*s maxpool. transpose=0 -> [b][pm][32]; 1 -> [b][32][n_pad] ----
__global__ void k_conv1x1_pool(const float* __restrict__ f1, const float* __restrict__ f2,
                               const float* __restrict__ w, const float* __restrict__ bias,
                               bf16* __restrict__ out, int H, int s, int np, int n_pad,
                               int transpose, int total) {
  __shared__ float ws_[32 * 64];
  __shared__ float bs_[32];
  for (int i = threadIdx.x; i < 32 * 64; i += blockDim.x) ws_[i] = w[i];
  if (threadIdx.x < 32) bs_[threadIdx.x] = bias[threadIdx.x];
  __syncthreads();
  int idx = blockIdx.x * blockDim.x + threadIdx.x;
  if (idx >= total) return;
  int n = np * np;
  int b = idx / n, pm = idx - b * n;
  int py = pm / np, px = pm - py * np;
  int N = H * H;
  float acc[32];
#pragma unroll
  for (int oc = 0; oc < 32; ++oc) acc[oc] = -3.0e38f;
  for (int dy = 0; dy < s; ++dy) {
    for (int dx = 0; dx < s; ++dx) {
      int p = (py * s + dy) * H + (px * s + dx);
      float xv[64];
#pragma unroll
      for (int c = 0; c < 32; ++c) xv[c] = f1[((size_t)(b * 32 + c)) * N + p];
#pragma unroll
      for (int c = 0; c < 32; ++c) xv[32 + c] = f2[((size_t)(b * 32 + c)) * N + p];
#pragma unroll
      for (int oc = 0; oc < 32; ++oc) {
        float a = bs_[oc];
#pragma unroll
        for (int c = 0; c < 64; ++c) a = fmaf(ws_[oc * 64 + c], xv[c], a);
        acc[oc] = fmaxf(acc[oc], a);
      }
    }
  }
  if (transpose) {
#pragma unroll
    for (int oc = 0; oc < 32; ++oc)
      out[((size_t)(b * 32 + oc)) * n_pad + pm] = (bf16)acc[oc];
  } else {
#pragma unroll
    for (int oc = 0; oc < 32; ++oc)
      out[((size_t)b * n_pad + pm) * 32 + oc] = (bf16)acc[oc];
  }
}

// ---- fused attention: y = softmax(theta^T phi) g^T.
// Two-pass softmax: pass 1 = lane-local row-max partials (reduction hoisted out
// of the loop), pass 2 = exact exp + WMMA accumulate, lane-local row sums.
// One wave per (b, 16-row tile of N). ----
__global__ void k_attn(const bf16* __restrict__ theta, const bf16* __restrict__ phi,
                       const bf16* __restrict__ gt, bf16* __restrict__ yout,
                       int N, int n, int n_pad, int nwaves) {
  __shared__ alignas(32) bf16 Pbuf[WPB][16 * 32];
  int wave = threadIdx.x >> 5, lane = threadIdx.x & 31;
  int gw = blockIdx.x * WPB + wave;
  if (gw >= nwaves) return;
  int tiles = N >> 4;
  int b = gw / tiles, it = gw - b * tiles;
  int lm = lane & 15, grp = lane >> 4;

  const bf16* trow = theta + ((size_t)b * N + it * 16 + lm) * 32;
  v16bf Af = load_fragA(trow, grp);
  const bf16* phiB = phi + (size_t)b * n_pad * 32;
  const bf16* gB = gt + (size_t)b * 32 * n_pad;
  bf16* Prow = &Pbuf[wave][0];
  const v8f zc = {0.f, 0.f, 0.f, 0.f, 0.f, 0.f, 0.f, 0.f};

  // ---- pass 1: per-row max, lane-local partials ----
  float mrow[8];
#pragma unroll
  for (int r = 0; r < 8; ++r) mrow[r] = -3.0e38f;
  for (int mb = 0; mb < n_pad; mb += 32) {
    v16bf B0 = load_fragB(phiB + (size_t)(mb + lm) * 32, grp);
    v16bf B1 = load_fragB(phiB + (size_t)(mb + 16 + lm) * 32, grp);
    v8f S0 = wmma_bf16(Af, B0, zc);
    v8f S1 = wmma_bf16(Af, B1, zc);
    bool ok0 = (mb + lm) < n, ok1 = (mb + 16 + lm) < n;
#pragma unroll
    for (int r = 0; r < 8; ++r) {
      if (ok0) mrow[r] = fmaxf(mrow[r], S0[r]);
      if (ok1) mrow[r] = fmaxf(mrow[r], S1[r]);
    }
  }
  // single cross-lane butterfly (masks 8..1 stay inside each 16-lane half)
#pragma unroll
  for (int off = 8; off; off >>= 1) {
#pragma unroll
    for (int r = 0; r < 8; ++r) mrow[r] = fmaxf(mrow[r], __shfl_xor(mrow[r], off, 32));
  }

  // ---- pass 2: exp with exact max, accumulate Y and lane-local row sums ----
  float lrow[8];
#pragma unroll
  for (int r = 0; r < 8; ++r) lrow[r] = 0.f;
  v8f Y0 = zc, Y1 = zc;
  for (int mb = 0; mb < n_pad; mb += 32) {
    v16bf B0 = load_fragB(phiB + (size_t)(mb + lm) * 32, grp);
    v16bf B1 = load_fragB(phiB + (size_t)(mb + 16 + lm) * 32, grp);
    v8f S0 = wmma_bf16(Af, B0, zc);
    v8f S1 = wmma_bf16(Af, B1, zc);
    bool ok0 = (mb + lm) < n, ok1 = (mb + 16 + lm) < n;
#pragma unroll
    for (int r = 0; r < 8; ++r) {
      float p0 = ok0 ? __expf(S0[r] - mrow[r]) : 0.f;
      float p1 = ok1 ? __expf(S1[r] - mrow[r]) : 0.f;
      lrow[r] += p0 + p1;
      // stage P into A-layout source: row (r+8*grp), cols lm and 16+lm
      Prow[(r + 8 * grp) * 32 + lm] = (bf16)p0;
      Prow[(r + 8 * grp) * 32 + 16 + lm] = (bf16)p1;
    }
    asm volatile("s_wait_dscnt 0" ::: "memory");  // wave-local LDS RAW fence
    v16bf Pf = load_fragA(Prow + lm * 32, grp);
    v16bf G0 = load_fragB(gB + (size_t)lm * n_pad + mb, grp);
    v16bf G1 = load_fragB(gB + (size_t)(16 + lm) * n_pad + mb, grp);
    Y0 = wmma_bf16(Pf, G0, Y0);
    Y1 = wmma_bf16(Pf, G1, Y1);
  }
#pragma unroll
  for (int off = 8; off; off >>= 1) {
#pragma unroll
    for (int r = 0; r < 8; ++r) lrow[r] += __shfl_xor(lrow[r], off, 32);
  }
  bf16* yr = yout + ((size_t)b * N + it * 16) * 32;
#pragma unroll
  for (int r = 0; r < 8; ++r) {
    float invl = 1.f / lrow[r];
    int row = r + 8 * grp;
    yr[(size_t)row * 32 + lm] = (bf16)(Y0[r] * invl);
    yr[(size_t)row * 32 + 16 + lm] = (bf16)(Y1[r] * invl);
  }
}

// ---- 1x1 conv W (32 -> 64) + bias + residual concat(fea1,fea2); z [b][p][64] ----
__global__ void k_convw_res(const bf16* __restrict__ y, const float* __restrict__ f1,
                            const float* __restrict__ f2, const float* __restrict__ w,
                            const float* __restrict__ bias, bf16* __restrict__ z,
                            int N, int total) {
  __shared__ float ws_[64 * 32];
  __shared__ float bs_[64];
  for (int i = threadIdx.x; i < 64 * 32; i += blockDim.x) ws_[i] = w[i];
  if (threadIdx.x < 64) bs_[threadIdx.x] = bias[threadIdx.x];
  __syncthreads();
  int idx = blockIdx.x * blockDim.x + threadIdx.x;
  if (idx >= total) return;
  int b = idx / N, p = idx - b * N;
  float yv[32];
#pragma unroll
  for (int c = 0; c < 32; ++c) yv[c] = (float)y[(size_t)idx * 32 + c];
  bf16* zr = z + (size_t)idx * 64;
  for (int oc = 0; oc < 64; ++oc) {
    float a = bs_[oc];
    const float* wr = &ws_[oc * 32];
#pragma unroll
    for (int c = 0; c < 32; ++c) a = fmaf(wr[c], yv[c], a);
    float res = (oc < 32) ? f1[((size_t)(b * 32 + oc)) * N + p]
                          : f2[((size_t)(b * 32 + oc - 32)) * N + p];
    zr[oc] = (bf16)(a + res);
  }
}

// ---- repack conv3x3 weights [32][64][3][3] -> bf16 [o][t*64+c] (t = ty*3+tx) ----
__global__ void k_prep_w2(const float* __restrict__ cw, bf16* __restrict__ w2, int total) {
  int idx = blockIdx.x * blockDim.x + threadIdx.x;
  if (idx >= total) return;
  int o = idx / 576, k = idx - o * 576;
  int t = k >> 6, c = k & 63;
  int ty = t / 3, tx = t - ty * 3;
  w2[idx] = (bf16)cw[(((size_t)o * 64 + c) * 3 + ty) * 3 + tx];
}

// ---- 3x3 conv (64 -> 32, pad 1) + BN via WMMA, one wave per 16-position tile ----
__global__ void k_conv3x3_bn(const bf16* __restrict__ z, const bf16* __restrict__ w2,
                             const float* __restrict__ gam, const float* __restrict__ bet,
                             const float* __restrict__ mea, const float* __restrict__ var,
                             float* __restrict__ out, int H, int nwaves) {
  int wave = threadIdx.x >> 5, lane = threadIdx.x & 31;
  int gw = blockIdx.x * WPB + wave;
  if (gw >= nwaves) return;
  int N = H * H;
  int tiles = N >> 4;
  int b = gw / tiles, it = gw - b * tiles;
  int lm = lane & 15, grp = lane >> 4;
  int p = it * 16 + lm;
  int py = p / H, px = p - py * H;
  v8f a0 = {0.f, 0.f, 0.f, 0.f, 0.f, 0.f, 0.f, 0.f};
  v8f a1 = {0.f, 0.f, 0.f, 0.f, 0.f, 0.f, 0.f, 0.f};
  const bf16* zb = z + (size_t)b * N * 64;
#pragma unroll
  for (int t = 0; t < 9; ++t) {
    int dy = t / 3 - 1, dx = t % 3 - 1;
    int ny = py + dy, nx = px + dx;
    bool ok = ((unsigned)ny < (unsigned)H) && ((unsigned)nx < (unsigned)H);
    const bf16* zr = zb + (size_t)(ny * H + nx) * 64;
#pragma unroll
    for (int cc = 0; cc < 2; ++cc) {
      v16bf A;
      if (ok) {
        A = load_fragA(zr + cc * 32, grp);
      } else {
#pragma unroll
        for (int i = 0; i < 16; ++i) A[i] = (bf16)0.f;
      }
      int kb = t * 64 + cc * 32;
      v16bf B0 = load_fragB(w2 + (size_t)lm * 576 + kb, grp);
      v16bf B1 = load_fragB(w2 + (size_t)(16 + lm) * 576 + kb, grp);
      a0 = wmma_bf16(A, B0, a0);
      a1 = wmma_bf16(A, B1, a1);
    }
  }
  float inv0 = gam[lm] * rsqrtf(var[lm] + EPSV);
  float add0 = bet[lm] - mea[lm] * inv0;
  float inv1 = gam[16 + lm] * rsqrtf(var[16 + lm] + EPSV);
  float add1 = bet[16 + lm] - mea[16 + lm] * inv1;
  float* ob = out + (size_t)b * 32 * N;
#pragma unroll
  for (int r = 0; r < 8; ++r) {
    int pos = it * 16 + r + 8 * grp;
    ob[(size_t)lm * N + pos] = a0[r] * inv0 + add0;
    ob[(size_t)(16 + lm) * N + pos] = a1[r] * inv1 + add1;
  }
}

extern "C" void kernel_launch(void* const* d_in, const int* in_sizes, int n_in,
                              void* d_out, int out_size, void* d_ws, size_t ws_size,
                              hipStream_t stream) {
  (void)in_sizes; (void)n_in; (void)out_size; (void)ws_size;
  const int B = 4;
  static const int Hs[3] = {96, 48, 24};
  static const int Ss[3] = {2, 4, 8};

  char* ws = (char*)d_ws;
  size_t off = 0;
  auto alloc = [&](size_t bytes) {
    void* pp = ws + off;
    off += (bytes + 255) & ~(size_t)255;
    return pp;
  };
  // sized for the largest level (H=96: N=9216, n_pad=2304); reused across levels
  bf16* theta = (bf16*)alloc((size_t)B * 9216 * 32 * sizeof(bf16));
  bf16* ybuf  = (bf16*)alloc((size_t)B * 9216 * 32 * sizeof(bf16));
  bf16* phi   = (bf16*)alloc((size_t)B * 2304 * 32 * sizeof(bf16));
  bf16* gtb   = (bf16*)alloc((size_t)B * 2304 * 32 * sizeof(bf16));
  bf16* zbuf  = (bf16*)alloc((size_t)B * 9216 * 64 * sizeof(bf16));
  bf16* w2    = (bf16*)alloc((size_t)32 * 576 * sizeof(bf16));

  float* outp = (float*)d_out;
  size_t out_off = 0;
  for (int lv = 0; lv < 3; ++lv) {
    int H = Hs[lv], s = Ss[lv];
    int N = H * H, np = H / s, n = np * np;
    int n_pad = (n + 31) & ~31;
    const float* f1 = (const float*)d_in[2 * lv];
    const float* f2 = (const float*)d_in[2 * lv + 1];
    int pb = 6 + lv * 13;
    const float* gww = (const float*)d_in[pb + 0];
    const float* gwb = (const float*)d_in[pb + 1];
    const float* tw  = (const float*)d_in[pb + 2];
    const float* tb  = (const float*)d_in[pb + 3];
    const float* pw  = (const float*)d_in[pb + 4];
    const float* pbv = (const float*)d_in[pb + 5];
    const float* ww  = (const float*)d_in[pb + 6];
    const float* wb  = (const float*)d_in[pb + 7];
    const float* cw  = (const float*)d_in[pb + 8];
    const float* bng = (const float*)d_in[pb + 9];
    const float* bnb = (const float*)d_in[pb + 10];
    const float* bnm = (const float*)d_in[pb + 11];
    const float* bnv = (const float*)d_in[pb + 12];

    int totN = B * N, totn = B * n;
    k_prep_w2<<<(32 * 576 + 255) / 256, 256, 0, stream>>>(cw, w2, 32 * 576);
    k_conv1x1<<<(totN + 255) / 256, 256, 0, stream>>>(f1, f2, tw, tb, theta, N, totN);
    k_conv1x1_pool<<<(totn + 255) / 256, 256, 0, stream>>>(f1, f2, pw, pbv, phi, H, s,
                                                           np, n_pad, 0, totn);
    k_conv1x1_pool<<<(totn + 255) / 256, 256, 0, stream>>>(f1, f2, gww, gwb, gtb, H, s,
                                                           np, n_pad, 1, totn);
    int nwaves = B * (N / 16);
    int nblk = (nwaves + WPB - 1) / WPB;
    k_attn<<<nblk, 32 * WPB, 0, stream>>>(theta, phi, gtb, ybuf, N, n, n_pad, nwaves);
    k_convw_res<<<(totN + 255) / 256, 256, 0, stream>>>(ybuf, f1, f2, ww, wb, zbuf, N, totN);
    k_conv3x3_bn<<<nblk, 32 * WPB, 0, stream>>>(zbuf, w2, bng, bnb, bnm, bnv,
                                                outp + out_off, H, nwaves);
    out_off += (size_t)B * 32 * N;
  }
}
